// StaticSparseAttention_14937896256261
// MI455X (gfx1250) — matrix operand order repaired and checked by reading, not compiled, not run on hardware
//
#include <hip/hip_runtime.h>

typedef __bf16 bf16_t;
typedef __attribute__((ext_vector_type(8)))  __bf16 v8bf;
typedef __attribute__((ext_vector_type(16))) __bf16 v16bf;
typedef __attribute__((ext_vector_type(8)))  float  v8f;

#define EMBED 768
#define HEADS 12
#define HDIM 64
#define L_PATCH 4096
#define S_TOK 4097           // L + 1 cls
#define S_PAD 4160           // 65 * 64 (macro-tile rows)
#define QKV_N 2304           // 3*EMBED
#define SCALE 0.125f         // 1/sqrt(64)

// ---------------------------------------------------------------------------
// f32 -> bf16 with zero-padding of rows >= validRows
// ---------------------------------------------------------------------------
__global__ void cvt_pad_bf16(bf16_t* __restrict__ dst, const float* __restrict__ src,
                             int validRows, int total, int cols) {
    int i = blockIdx.x * 256 + threadIdx.x;
    if (i >= total) return;
    int row = i / cols;
    dst[i] = (row < validRows) ? (bf16_t)src[i] : (bf16_t)0.0f;
}

__global__ void zero_bf16(bf16_t* __restrict__ dst, int n) {
    int i = blockIdx.x * 256 + threadIdx.x;
    if (i < n) dst[i] = (bf16_t)0.0f;
}

// ---------------------------------------------------------------------------
// Register-blocked WMMA bf16 GEMM:  C[m,n] = sum_k A[m,k] * W[n,k] + bias[n]
// A: bf16 row-major [MtilesB*64, K] (pad rows zeroed), W: bf16 row-major [N, K]
// One wave32 computes a 64x32 C macro-tile = 4x2 grid of 16x16 WMMA tiles:
// 8 v_wmma_f32_16x16x32_bf16 per K-step, A frags reused 2x, B frags 4x.
// ---------------------------------------------------------------------------
__global__ void wmma_gemm_bias(const bf16_t* __restrict__ A,
                               const bf16_t* __restrict__ W,
                               const float* __restrict__ bias,
                               float* __restrict__ C,
                               int Mvalid, int MtilesB, int NtilesB, int N, int K) {
    int wave = (int)((blockIdx.x * blockDim.x + threadIdx.x) >> 5);
    if (wave >= MtilesB * NtilesB) return;        // uniform per wave
    int tm = wave / NtilesB;
    int tn = wave - tm * NtilesB;
    int lane = threadIdx.x & 31;
    int half = lane >> 4;                         // 0: lanes 0-15, 1: lanes 16-31
    int l15  = lane & 15;

    const bf16_t* Arow[4];
    const bf16_t* Wrow[2];
    #pragma unroll
    for (int mi = 0; mi < 4; ++mi)
        Arow[mi] = A + (size_t)(tm * 64 + mi * 16 + l15) * K;   // lane = M
    #pragma unroll
    for (int ni = 0; ni < 2; ++ni)
        Wrow[ni] = W + (size_t)(tn * 32 + ni * 16 + l15) * K;   // lane = N

    const int kgA = half ? 16 : 0;                // A: e0-15 -> K{0-15|16-31}
    const int kgB = half ? 16 : 0;                // B: e0-15 -> K{0-15|16-31}

    v8f acc[4][2] = {};
    for (int k0 = 0; k0 < K; k0 += 32) {
        v16bf a[4], b[2];
        #pragma unroll
        for (int mi = 0; mi < 4; ++mi) {
            v8bf alo = *reinterpret_cast<const v8bf*>(Arow[mi] + k0 + kgA);
            v8bf ahi = *reinterpret_cast<const v8bf*>(Arow[mi] + k0 + 8 + kgA);
            a[mi] = __builtin_shufflevector(alo, ahi,
                     0,1,2,3,4,5,6,7,8,9,10,11,12,13,14,15);
        }
        #pragma unroll
        for (int ni = 0; ni < 2; ++ni)
            b[ni] = *reinterpret_cast<const v16bf*>(Wrow[ni] + k0 + kgB);
        #pragma unroll
        for (int mi = 0; mi < 4; ++mi)
            #pragma unroll
            for (int ni = 0; ni < 2; ++ni)
                acc[mi][ni] = __builtin_amdgcn_wmma_f32_16x16x32_bf16(
                    false, a[mi], false, b[ni], (short)0, acc[mi][ni], false, false);
    }

    #pragma unroll
    for (int ni = 0; ni < 2; ++ni) {
        int n = tn * 32 + ni * 16 + l15;
        float bv = bias[n];
        #pragma unroll
        for (int mi = 0; mi < 4; ++mi) {
            #pragma unroll
            for (int r = 0; r < 8; ++r) {
                int m = tm * 64 + mi * 16 + r + half * 8;  // C: VGPR r -> M=r / M=8+r
                if (m < Mvalid) C[(size_t)m * N + n] = acc[mi][ni][r] + bv;
            }
        }
    }
}

// ---------------------------------------------------------------------------
// RoPE2D applied in-place to q,k of patch tokens. qkv layout: [S][3][12][64].
// Thread per (patch l, head h, rotation pair p in 0..31).
// ---------------------------------------------------------------------------
__global__ void rope_kernel(float* __restrict__ qkv, const int* __restrict__ coords) {
    int idx = blockIdx.x * 256 + threadIdx.x;
    if (idx >= L_PATCH * HEADS * 32) return;
    int p = idx & 31;
    int h = (idx >> 5) % HEADS;
    int l = idx / (32 * HEADS);
    int s = l + 1;
    float cx = (float)coords[2 * l]     * 1e-5f;
    float cy = (float)coords[2 * l + 1] * 1e-5f;
    float invf = __powf(10000.0f, -(float)(p & 15) * (1.0f / 16.0f));
    float ang = (p < 16 ? cx : cy) * invf;
    float c = __cosf(ang), sn = __sinf(ang);
    size_t bq = (size_t)s * QKV_N + h * HDIM + 2 * p;
    size_t bk = bq + EMBED;
    float q0 = qkv[bq], q1 = qkv[bq + 1];
    qkv[bq]     = q0 * c - q1 * sn;
    qkv[bq + 1] = q1 * c + q0 * sn;
    float k0 = qkv[bk], k1 = qkv[bk + 1];
    qkv[bk]     = k0 * c - k1 * sn;
    qkv[bk + 1] = k1 * c + k0 * sn;
}

// ---------------------------------------------------------------------------
// CLS token attends over all S keys. One 256-thread block per head.
// Output written bf16 into attnout row 0 (head-major layout).
// ---------------------------------------------------------------------------
__global__ void cls_attn(const float* __restrict__ qkv, bf16_t* __restrict__ attnout) {
    const int h = blockIdx.x;
    const int tid = threadIdx.x;
    __shared__ float sc[S_TOK];
    __shared__ float qs[HDIM];
    __shared__ float red[256];
    __shared__ float outacc[HDIM];
    if (tid < HDIM) { qs[tid] = qkv[(size_t)h * HDIM + tid]; outacc[tid] = 0.0f; }
    __syncthreads();

    float lmax = -1e30f;
    for (int s = tid; s < S_TOK; s += 256) {
        const float* kp = qkv + (size_t)s * QKV_N + EMBED + h * HDIM;
        float a = 0.0f;
        for (int d = 0; d < HDIM; ++d) a += qs[d] * kp[d];
        a *= SCALE;
        sc[s] = a;
        lmax = fmaxf(lmax, a);
    }
    red[tid] = lmax; __syncthreads();
    for (int off = 128; off > 0; off >>= 1) {
        if (tid < off) red[tid] = fmaxf(red[tid], red[tid + off]);
        __syncthreads();
    }
    float gmax = red[0];
    __syncthreads();

    float lsum = 0.0f;
    for (int s = tid; s < S_TOK; s += 256) {
        float p = __expf(sc[s] - gmax);
        sc[s] = p; lsum += p;
    }
    red[tid] = lsum; __syncthreads();
    for (int off = 128; off > 0; off >>= 1) {
        if (tid < off) red[tid] += red[tid + off];
        __syncthreads();
    }
    float inv = 1.0f / red[0];

    float local[HDIM];
    #pragma unroll
    for (int d = 0; d < HDIM; ++d) local[d] = 0.0f;
    for (int s = tid; s < S_TOK; s += 256) {
        float p = sc[s];
        const float* vp = qkv + (size_t)s * QKV_N + 2 * EMBED + h * HDIM;
        for (int d = 0; d < HDIM; ++d) local[d] += p * vp[d];
    }
    for (int d = 0; d < HDIM; ++d) atomicAdd(&outacc[d], local[d]);
    __syncthreads();
    if (tid < HDIM) attnout[h * HDIM + tid] = (bf16_t)(outacc[tid] * inv);
}

// ---------------------------------------------------------------------------
// Patch windowed attention (33 dilated keys + cls). One wave32 per (h, l).
// ---------------------------------------------------------------------------
__global__ void patch_attn(const float* __restrict__ qkv, bf16_t* __restrict__ attnout) {
    int gwave = (int)((blockIdx.x * blockDim.x + threadIdx.x) >> 5);
    int lane = threadIdx.x & 31;
    int wib = threadIdx.x >> 5;
    int h = gwave >> 12;          // / 4096
    int l = gwave & 4095;
    int s = l + 1;
    __shared__ float qsh[4][HDIM];
    __shared__ float psh[4][34];

    const float* qp = qkv + (size_t)s * QKV_N + h * HDIM;
    qsh[wib][lane] = qp[lane];
    qsh[wib][lane + 32] = qp[lane + 32];
    __syncthreads();

    auto key_row = [&](int j, bool& valid) -> int {
        if (j == 0) { valid = true; return 0; }        // cls
        int ki = l + (j - 17) * 4;                     // (w-16)*DIL, w=j-1
        valid = (ki >= 0) && (ki < L_PATCH);
        return ki + 1;
    };

    float sc0 = -1e30f, sc1 = -1e30f;
    {
        bool v0; int r0 = key_row(lane, v0);
        if (v0) {
            const float* kp = qkv + (size_t)r0 * QKV_N + EMBED + h * HDIM;
            float a = 0.0f;
            for (int d = 0; d < HDIM; ++d) a += qsh[wib][d] * kp[d];
            sc0 = a * SCALE;
        }
        int j1 = lane + 32;
        if (j1 < 34) {
            bool v1; int r1 = key_row(j1, v1);
            if (v1) {
                const float* kp = qkv + (size_t)r1 * QKV_N + EMBED + h * HDIM;
                float a = 0.0f;
                for (int d = 0; d < HDIM; ++d) a += qsh[wib][d] * kp[d];
                sc1 = a * SCALE;
            }
        }
    }
    float m = fmaxf(sc0, sc1);
    for (int off = 16; off >= 1; off >>= 1) m = fmaxf(m, __shfl_xor(m, off));
    float p0 = __expf(sc0 - m);
    float p1 = (lane + 32 < 34) ? __expf(sc1 - m) : 0.0f;
    float sum = p0 + p1;
    for (int off = 16; off >= 1; off >>= 1) sum += __shfl_xor(sum, off);
    float inv = 1.0f / sum;
    psh[wib][lane] = p0 * inv;
    if (lane < 2) psh[wib][lane + 32] = p1 * inv;
    __syncthreads();

    float o0 = 0.0f, o1 = 0.0f;
    for (int j = 0; j < 34; ++j) {
        bool v; int r = key_row(j, v);
        if (!v) continue;
        float p = psh[wib][j];
        const float* vp = qkv + (size_t)r * QKV_N + 2 * EMBED + h * HDIM;
        o0 += p * vp[lane];
        o1 += p * vp[lane + 32];
    }
    size_t ob = (size_t)s * EMBED + h * HDIM;
    attnout[ob + lane]      = (bf16_t)o0;
    attnout[ob + lane + 32] = (bf16_t)o1;
}

// ---------------------------------------------------------------------------
extern "C" void kernel_launch(void* const* d_in, const int* in_sizes, int n_in,
                              void* d_out, int out_size, void* d_ws, size_t ws_size,
                              hipStream_t stream) {
    const float* x      = (const float*)d_in[0];
    const int*   coords = (const int*)d_in[1];
    const float* qkv_w  = (const float*)d_in[2];
    const float* qkv_b  = (const float*)d_in[3];
    const float* out_w  = (const float*)d_in[4];
    const float* out_b  = (const float*)d_in[5];
    float* out = (float*)d_out;

    char* ws = (char*)d_ws;
    constexpr size_t SZ_XB   = (size_t)S_PAD * EMBED * 2;       // x bf16 padded
    constexpr size_t SZ_WQKV = (size_t)QKV_N * EMBED * 2;       // qkv_w bf16
    constexpr size_t SZ_WOUT = (size_t)EMBED * EMBED * 2;       // out_w bf16
    constexpr size_t SZ_QKV  = (size_t)S_TOK * QKV_N * 4;       // qkv f32
    size_t o = 0;
    bf16_t* xb   = (bf16_t*)(ws + o); o += SZ_XB;
    bf16_t* wqkv = (bf16_t*)(ws + o); o += SZ_WQKV;
    bf16_t* wout = (bf16_t*)(ws + o); o += SZ_WOUT;
    float*  qkv  = (float*) (ws + o); o += SZ_QKV;
    bf16_t* attn = (bf16_t*)(ws + o);                           // [S_PAD][768] bf16

    // 1) precision conversion (RNE) + padding
    {
        int n = S_PAD * EMBED;
        cvt_pad_bf16<<<(n + 255) / 256, 256, 0, stream>>>(xb, x, S_TOK, n, EMBED);
    }
    {
        int n = QKV_N * EMBED;
        cvt_pad_bf16<<<(n + 255) / 256, 256, 0, stream>>>(wqkv, qkv_w, QKV_N, n, EMBED);
    }
    {
        int n = EMBED * EMBED;
        cvt_pad_bf16<<<(n + 255) / 256, 256, 0, stream>>>(wout, out_w, EMBED, n, EMBED);
    }
    {
        int n = (S_PAD - S_TOK) * EMBED;   // zero attn pad rows
        zero_bf16<<<(n + 255) / 256, 256, 0, stream>>>(attn + (size_t)S_TOK * EMBED, n);
    }

    // 2) QKV projection GEMM (WMMA bf16): [4160,768] x [2304,768]^T
    {
        int Mtb = S_PAD / 64, Ntb = QKV_N / 32;      // 65 x 72 macro-tiles
        int waves = Mtb * Ntb;
        wmma_gemm_bias<<<(waves + 3) / 4, 128, 0, stream>>>(
            xb, wqkv, qkv_b, qkv, S_TOK, Mtb, Ntb, QKV_N, EMBED);
    }

    // 3) RoPE2D on patch q,k
    {
        int n = L_PATCH * HEADS * 32;
        rope_kernel<<<(n + 255) / 256, 256, 0, stream>>>(qkv, coords);
    }

    // 4) attention
    cls_attn<<<HEADS, 256, 0, stream>>>(qkv, attn);
    {
        int waves = HEADS * L_PATCH;                 // 49152
        patch_attn<<<waves / 4, 128, 0, stream>>>(qkv, attn);
    }

    // 5) output projection GEMM (WMMA bf16): [4160,768] x [768,768]^T
    {
        int Mtb = S_PAD / 64, Ntb = EMBED / 32;      // 65 x 24 macro-tiles
        int waves = Mtb * Ntb;
        wmma_gemm_bias<<<(waves + 3) / 4, 128, 0, stream>>>(
            attn, wout, out_b, out, S_TOK, Mtb, Ntb, EMBED, EMBED);
    }
}
